// GPSConv_73340861546553
// MI455X (gfx1250) — compile-verified, matrix operand order checked by reading, not run on hardware
//
#include <hip/hip_runtime.h>
#include <hip/hip_bf16.h>
#include <math.h>

// Problem constants (match reference)
#define NGRAPH 512
#define NPG 32
#define NNODE (NGRAPH * NPG)   // 16384
#define CDIM 128
#define EDIMK 16
#define NEDGE (NGRAPH * 256)   // 131072
#define EDCH 256               // ED
#define DSTATE 16
#define DTR 8
#define DCONV 4
#define HID 512
#define EPSF 1e-5f

typedef __attribute__((ext_vector_type(16))) _Float16 v16h;
typedef __attribute__((ext_vector_type(8)))  float    v8f;

// ---------------------------------------------------------------------------
// Fragment builders (CDNA5 ISA 7.12.2 layouts: 16-bit A 16x32, B 32x16)
// ---------------------------------------------------------------------------
__device__ __forceinline__ v16h load_a_frag(const float* __restrict__ A,
                                            long aRow, int k0, int hi)
{
    const float4 a0 = *(const float4*)(A + aRow + k0 + hi * 8);
    const float4 a1 = *(const float4*)(A + aRow + k0 + hi * 8 + 4);
    const float4 a2 = *(const float4*)(A + aRow + k0 + 16 + hi * 8);
    const float4 a3 = *(const float4*)(A + aRow + k0 + 16 + hi * 8 + 4);
    v16h a;
    a[0]  = (_Float16)a0.x; a[1]  = (_Float16)a0.y;
    a[2]  = (_Float16)a0.z; a[3]  = (_Float16)a0.w;
    a[4]  = (_Float16)a1.x; a[5]  = (_Float16)a1.y;
    a[6]  = (_Float16)a1.z; a[7]  = (_Float16)a1.w;
    a[8]  = (_Float16)a2.x; a[9]  = (_Float16)a2.y;
    a[10] = (_Float16)a2.z; a[11] = (_Float16)a2.w;
    a[12] = (_Float16)a3.x; a[13] = (_Float16)a3.y;
    a[14] = (_Float16)a3.z; a[15] = (_Float16)a3.w;
    return a;
}

__device__ __forceinline__ v16h load_b_frag(const float* __restrict__ bCol,
                                            int k0, int hi, int Nn)
{
    v16h b;
#pragma unroll
    for (int g = 0; g < 2; ++g)
#pragma unroll
        for (int w = 0; w < 8; ++w)
            b[g * 8 + w] = (_Float16)bCol[(long)(k0 + g * 16 + hi * 8 + w) * Nn];
    return b;
}

// ---------------------------------------------------------------------------
// FAST WMMA GEMM: requires K % 32 == 0, N % 128 == 0, M % 32 == 0.
// Block = 128 threads (4 waves) covering a 32(M) x 128(N) super-tile.
// Each wave computes a 32x32 output = 2x2 WMMA tiles: each A fragment is
// reused by 2 WMMAs and each B fragment by 2 WMMAs (8 FLOP/byte from L2,
// vs 4 for a 1x1 tile). All loads unconditional (no EXEC divergence).
// ---------------------------------------------------------------------------
__global__ __launch_bounds__(128)
void wmma_gemm_fast_kernel(const float* __restrict__ A, const float* __restrict__ B,
                           const float* __restrict__ bias, float* __restrict__ C,
                           int M, int Nn, int K)
{
    const int lane = threadIdx.x & 31;
    const int wib  = threadIdx.x >> 5;
    const int nStrip = Nn >> 7;                       // N/128 strips
    const int bm = (blockIdx.x / nStrip) << 5;        // 32-row band
    const int tn = ((blockIdx.x % nStrip) << 7) + (wib << 5);  // 32-col band
    const int low = lane & 15;
    const int hi  = lane >> 4;
    const long aRow0 = (long)(bm + low) * K;
    const long aRow1 = (long)(bm + 16 + low) * K;
    const float* __restrict__ bCol0 = B + tn + low;
    const float* __restrict__ bCol1 = B + tn + 16 + low;

    v8f acc00 = {}, acc01 = {}, acc10 = {}, acc11 = {};
    for (int k0 = 0; k0 < K; k0 += 32) {
        const v16h a0 = load_a_frag(A, aRow0, k0, hi);
        const v16h a1 = load_a_frag(A, aRow1, k0, hi);
        const v16h b0 = load_b_frag(bCol0, k0, hi, Nn);
        const v16h b1 = load_b_frag(bCol1, k0, hi, Nn);
        acc00 = __builtin_amdgcn_wmma_f32_16x16x32_f16(false, a0, false, b0,
                                                       (short)0, acc00, false, false);
        acc01 = __builtin_amdgcn_wmma_f32_16x16x32_f16(false, a0, false, b1,
                                                       (short)0, acc01, false, false);
        acc10 = __builtin_amdgcn_wmma_f32_16x16x32_f16(false, a1, false, b0,
                                                       (short)0, acc10, false, false);
        acc11 = __builtin_amdgcn_wmma_f32_16x16x32_f16(false, a1, false, b1,
                                                       (short)0, acc11, false, false);
    }

    const float bb0 = bias ? bias[tn + low] : 0.0f;
    const float bb1 = bias ? bias[tn + 16 + low] : 0.0f;
#pragma unroll
    for (int r = 0; r < 8; ++r) {
        const long row0 = (long)(bm + r + 8 * hi) * Nn;
        const long row1 = (long)(bm + 16 + r + 8 * hi) * Nn;
        C[row0 + tn + low]      = acc00[r] + bb0;
        C[row0 + tn + 16 + low] = acc01[r] + bb1;
        C[row1 + tn + low]      = acc10[r] + bb0;
        C[row1 + tn + 16 + low] = acc11[r] + bb1;
    }
}

// ---------------------------------------------------------------------------
// SAFE WMMA GEMM: arbitrary N, K (M % 16 == 0). Loads are UNCONDITIONAL with
// clamped addresses; masking is a value-select afterwards (v_cndmask), so no
// EXEC-mask branching in the inner loop. One wave per 16x16 tile.
// ---------------------------------------------------------------------------
__global__ __launch_bounds__(128)
void wmma_gemm_safe_kernel(const float* __restrict__ A, const float* __restrict__ B,
                           const float* __restrict__ bias, float* __restrict__ C,
                           int M, int Nn, int K)
{
    const int lane = threadIdx.x & 31;
    const int wib  = threadIdx.x >> 5;
    const int ntN  = (Nn + 15) >> 4;
    const long tile = (long)blockIdx.x * 4 + wib;
    if (tile >= (long)(M >> 4) * ntN) return;       // wave-uniform

    const int tm = (int)(tile / ntN) << 4;
    const int tn = (int)(tile % ntN) << 4;
    const int low = lane & 15;
    const int hi  = lane >> 4;
    const int rowA  = tm + low;                     // always < M (M%16==0)
    const int colB  = tn + low;
    const int colBc = colB < Nn ? colB : Nn - 1;    // clamped, always valid

    v8f acc = {};
    for (int k0 = 0; k0 < K; k0 += 32) {
        v16h a, b;
#pragma unroll
        for (int g = 0; g < 2; ++g)
#pragma unroll
            for (int w = 0; w < 8; ++w) {
                const int kk = k0 + g * 16 + hi * 8 + w;
                const int kc = kk < K ? kk : K - 1;  // clamped address
                float av = A[(long)rowA * K + kc];   // unconditional load
                float bv = B[(long)kc * Nn + colBc]; // unconditional load
                av = (kk < K) ? av : 0.0f;                       // value select
                bv = (kk < K && colB < Nn) ? bv : 0.0f;          // value select
                a[g * 8 + w] = (_Float16)av;
                b[g * 8 + w] = (_Float16)bv;
            }
        acc = __builtin_amdgcn_wmma_f32_16x16x32_f16(
            false, a, false, b, (short)0, acc, false, false);
    }

    if (colB < Nn) {
        const float bb = bias ? bias[colB] : 0.0f;
#pragma unroll
        for (int r = 0; r < 8; ++r)
            C[(long)(tm + r + 8 * hi) * Nn + colB] = acc[r] + bb;
    }
}

// ---------------------------------------------------------------------------
// Attention / elementwise kernels
// ---------------------------------------------------------------------------
__device__ __forceinline__ float wave_reduce_add(float v) {
#pragma unroll
    for (int off = 16; off; off >>= 1) v += __shfl_xor(v, off, 32);
    return v;
}

__device__ __forceinline__ void atomicMaxF(float* addr, float val) {
    int* ai = (int*)addr;
    int old = __float_as_int(*addr);
    while (__int_as_float(old) < val) {
        int assumed = old;
        old = atomicCAS(ai, assumed, __float_as_int(val));
        if (old == assumed) break;
    }
}

__global__ void init_attn_kernel(float* m, float* s, float* agg) {
    long i = (long)blockIdx.x * blockDim.x + threadIdx.x;
    if (i < NNODE) { m[i] = -INFINITY; s[i] = 0.0f; }
    if (i < (long)NNODE * CDIM) agg[i] = 0.0f;
}

// one wave per edge: logit = dot(q[dst], k[src]+e) / sqrt(C)
__global__ void edge_logit_kernel(const float* __restrict__ q, const float* __restrict__ k,
                                  const float* __restrict__ e, const int* __restrict__ ei,
                                  float* __restrict__ logit)
{
    long wid = ((long)blockIdx.x * blockDim.x + threadIdx.x) >> 5;
    int lane = threadIdx.x & 31;
    if (wid >= NEDGE) return;
    int src = ei[wid];
    int dst = ei[NEDGE + wid];
    float p = 0.0f;
#pragma unroll
    for (int j = 0; j < CDIM / 32; ++j) {
        int c = lane + 32 * j;
        p += q[(long)dst * CDIM + c] * (k[(long)src * CDIM + c] + e[wid * CDIM + c]);
    }
    p = wave_reduce_add(p);
    if (lane == 0) logit[wid] = p * 0.08838834764831845f;   // 1/sqrt(128)
}

__global__ void edge_max_kernel(const float* __restrict__ logit, const int* __restrict__ ei,
                                float* m)
{
    long i = (long)blockIdx.x * blockDim.x + threadIdx.x;
    if (i >= NEDGE) return;
    atomicMaxF(&m[ei[NEDGE + i]], logit[i]);
}

__global__ void edge_exp_kernel(const float* __restrict__ logit, const int* __restrict__ ei,
                                const float* __restrict__ m, float* __restrict__ ea, float* s)
{
    long i = (long)blockIdx.x * blockDim.x + threadIdx.x;
    if (i >= NEDGE) return;
    int d = ei[NEDGE + i];
    float v = __expf(logit[i] - m[d]);
    ea[i] = v;
    atomicAdd(&s[d], v);
}

__global__ void edge_agg_kernel(const float* __restrict__ v, const float* __restrict__ e,
                                const float* __restrict__ ea, const float* __restrict__ s,
                                const int* __restrict__ ei, float* agg)
{
    long idx = (long)blockIdx.x * blockDim.x + threadIdx.x;   // over NEDGE*CDIM
    if (idx >= (long)NEDGE * CDIM) return;
    long edge = idx >> 7;          // /128
    int  c    = (int)(idx & 127);
    int src = ei[edge];
    int dst = ei[NEDGE + edge];
    float alpha = ea[edge] / (s[dst] + 1e-16f);
    atomicAdd(&agg[(long)dst * CDIM + c], (v[(long)src * CDIM + c] + e[idx]) * alpha);
}

// wave per node: beta gate, writes mixed result in-place into agg
__global__ void beta_mix_kernel(float* agg, const float* __restrict__ xr,
                                const float* __restrict__ wb)
{
    long n = ((long)blockIdx.x * blockDim.x + threadIdx.x) >> 5;
    int lane = threadIdx.x & 31;
    if (n >= NNODE) return;
    float p = 0.0f;
#pragma unroll
    for (int j = 0; j < CDIM / 32; ++j) {
        int c = lane + 32 * j;
        float a = agg[n * CDIM + c], r = xr[n * CDIM + c];
        p += a * wb[c] + r * wb[CDIM + c] + (a - r) * wb[2 * CDIM + c];
    }
    p = wave_reduce_add(p);
    float beta = 1.0f / (1.0f + __expf(-p));
#pragma unroll
    for (int j = 0; j < CDIM / 32; ++j) {
        int c = lane + 32 * j;
        float a = agg[n * CDIM + c], r = xr[n * CDIM + c];
        agg[n * CDIM + c] = beta * r + (1.0f - beta) * a;
    }
}

// GraphNorm: block per graph (NG blocks), 128 threads = one per feature.
__global__ __launch_bounds__(CDIM)
void graph_norm_kernel(const float* __restrict__ hin, const float* __restrict__ addin,
                       const float* __restrict__ out_add, float* __restrict__ out,
                       const float* __restrict__ gnw, const float* __restrict__ gnb,
                       const float* __restrict__ gnms)
{
    int g = blockIdx.x;
    int c = threadIdx.x;
    float col[NPG];
    float mean = 0.0f;
#pragma unroll
    for (int t = 0; t < NPG; ++t) {
        long idx = ((long)g * NPG + t) * CDIM + c;
        float v = hin[idx] + (addin ? addin[idx] : 0.0f);
        col[t] = v;
        mean += v;
    }
    mean *= (1.0f / NPG);
    float ms = gnms[c];
    float var = 0.0f;
#pragma unroll
    for (int t = 0; t < NPG; ++t) {
        float hc = col[t] - mean * ms;
        col[t] = hc;
        var += hc * hc;
    }
    var *= (1.0f / NPG);
    float inv = rsqrtf(var + EPSF);
    float w = gnw[c], b = gnb[c];
#pragma unroll
    for (int t = 0; t < NPG; ++t) {
        long idx = ((long)g * NPG + t) * CDIM + c;
        float r = w * col[t] * inv + b;
        if (out_add) r += out_add[idx];
        out[idx] = r;
    }
}

// ---------------------------------------------------------------------------
// Mamba kernels. Backward direction runs in "flipped" (conv) coordinates.
// ---------------------------------------------------------------------------
__global__ void mamba_conv_kernel(const float* __restrict__ xz, const float* __restrict__ cw,
                                  const float* __restrict__ cb, float* __restrict__ xiout,
                                  int rev)
{
    long idx = (long)blockIdx.x * blockDim.x + threadIdx.x;   // over NNODE*EDCH
    if (idx >= (long)NNODE * EDCH) return;
    int e = (int)(idx % EDCH);
    long node = idx / EDCH;
    int g = (int)(node / NPG);
    int t = (int)(node % NPG);
    float acc = cb[e];
#pragma unroll
    for (int j = 0; j < DCONV; ++j) {
        int s = t - (DCONV - 1) + j;
        int sc = s >= 0 ? s : 0;                          // clamped address
        int sr = rev ? (NPG - 1 - sc) : sc;
        float v = xz[((long)g * NPG + sr) * (2 * EDCH) + e];  // unconditional
        v = (s >= 0) ? v : 0.0f;                          // value select
        acc += cw[j * EDCH + e] * v;
    }
    xiout[idx] = acc / (1.0f + __expf(-acc));   // silu
}

__global__ void mamba_delta_kernel(const float* __restrict__ dbc, const float* __restrict__ dtw,
                                   const float* __restrict__ dtb, float* __restrict__ delta)
{
    long idx = (long)blockIdx.x * blockDim.x + threadIdx.x;   // over NNODE*EDCH
    if (idx >= (long)NNODE * EDCH) return;
    int e = (int)(idx % EDCH);
    long node = idx / EDCH;
    float acc = dtb[e];
#pragma unroll
    for (int r = 0; r < DTR; ++r) acc += dbc[node * 40 + r] * dtw[r * EDCH + e];
    delta[idx] = (acc > 20.0f) ? acc : log1pf(__expf(acc));   // softplus
}

// one thread per (graph, channel); 16-wide SSM state in registers
__global__ void mamba_scan_kernel(const float* __restrict__ xi, const float* __restrict__ delta,
                                  const float* __restrict__ dbc, const float* __restrict__ xz,
                                  const float* __restrict__ Alog, const float* __restrict__ Dp,
                                  float* __restrict__ yz, int rev)
{
    long idx = (long)blockIdx.x * blockDim.x + threadIdx.x;   // over NGRAPH*EDCH
    if (idx >= (long)NGRAPH * EDCH) return;
    int e = (int)(idx % EDCH);
    int g = (int)(idx / EDCH);
    float A[DSTATE], h[DSTATE];
#pragma unroll
    for (int s = 0; s < DSTATE; ++s) { A[s] = -__expf(Alog[e * DSTATE + s]); h[s] = 0.0f; }
    float Dv = Dp[e];
    for (int t = 0; t < NPG; ++t) {
        long node = (long)g * NPG + t;
        float d  = delta[node * EDCH + e];
        float xv = xi[node * EDCH + e];
        float dx = d * xv;
        const float* row = dbc + node * 40;
        float y = 0.0f;
#pragma unroll
        for (int s = 0; s < DSTATE; ++s) {
            float dA = __expf(d * A[s]);
            h[s] = dA * h[s] + dx * row[8 + s];
            y += h[s] * row[24 + s];
        }
        y += Dv * xv;
        int zt = rev ? (NPG - 1 - t) : t;
        float z = xz[((long)g * NPG + zt) * (2 * EDCH) + EDCH + e];
        yz[node * EDCH + e] = y * (z / (1.0f + __expf(-z)));
    }
}

// wave per node: acc = x + mix_f + flip(mix_b); LayerNorm over C
__global__ void combine_ln_kernel(const float* __restrict__ x, const float* __restrict__ mf,
                                  const float* __restrict__ mb, const float* __restrict__ lng,
                                  const float* __restrict__ lnb, float* __restrict__ out)
{
    long n = ((long)blockIdx.x * blockDim.x + threadIdx.x) >> 5;
    int lane = threadIdx.x & 31;
    if (n >= NNODE) return;
    int g = (int)(n >> 5), t = (int)(n & 31);
    long nb = ((long)g << 5) + (31 - t);
    float vals[CDIM / 32];
    float sum = 0.0f, sq = 0.0f;
#pragma unroll
    for (int j = 0; j < CDIM / 32; ++j) {
        int c = lane + 32 * j;
        float v = x[n * CDIM + c] + mf[n * CDIM + c] + mb[nb * CDIM + c];
        vals[j] = v; sum += v; sq += v * v;
    }
    sum = wave_reduce_add(sum);
    sq  = wave_reduce_add(sq);
    float mean = sum * (1.0f / CDIM);
    float var  = sq * (1.0f / CDIM) - mean * mean;
    float inv  = rsqrtf(var + EPSF);
#pragma unroll
    for (int j = 0; j < CDIM / 32; ++j) {
        int c = lane + 32 * j;
        out[n * CDIM + c] = (vals[j] - mean) * inv * lng[c] + lnb[c];
    }
}

__global__ void mlp_gate_kernel(float* a1, const float* __restrict__ a2) {
    long i = (long)blockIdx.x * blockDim.x + threadIdx.x;   // over NNODE*HID
    if (i >= (long)NNODE * HID) return;
    float a = a1[i];
    a1[i] = (a / (1.0f + __expf(-a))) * a2[i];
}

// ---------------------------------------------------------------------------
// Host orchestration
// ---------------------------------------------------------------------------
static inline void launch_gemm(const float* A, const float* B, const float* bias,
                               float* C, int M, int Nn, int K, hipStream_t st)
{
    if ((K % 32 == 0) && (Nn % 128 == 0) && (M % 32 == 0)) {
        int blocks = (M / 32) * (Nn / 128);
        wmma_gemm_fast_kernel<<<blocks, 128, 0, st>>>(A, B, bias, C, M, Nn, K);
    } else {
        long tiles = (long)((M + 15) / 16) * ((Nn + 15) / 16);
        int blocks = (int)((tiles + 3) / 4);
        wmma_gemm_safe_kernel<<<blocks, 128, 0, st>>>(A, B, bias, C, M, Nn, K);
    }
}

extern "C" void kernel_launch(void* const* d_in, const int* in_sizes, int n_in,
                              void* d_out, int out_size, void* d_ws, size_t ws_size,
                              hipStream_t stream)
{
    (void)in_sizes; (void)n_in; (void)out_size; (void)ws_size;

    const float* x     = (const float*)d_in[0];
    const int*   ei    = (const int*)d_in[1];
    const float* eattr = (const float*)d_in[2];
    // d_in[3] = batch (implicit: node/32)
    const float* wq = (const float*)d_in[4],  *bq = (const float*)d_in[5];
    const float* wk = (const float*)d_in[6],  *bk = (const float*)d_in[7];
    const float* wv = (const float*)d_in[8],  *bv = (const float*)d_in[9];
    const float* we = (const float*)d_in[10], *be = (const float*)d_in[11];
    const float* wskip = (const float*)d_in[12], *bskip = (const float*)d_in[13];
    const float* wbeta = (const float*)d_in[14];
    const float* wlin  = (const float*)d_in[15], *blin = (const float*)d_in[16];
    const float* gnw = (const float*)d_in[17], *gnb = (const float*)d_in[18],
               * gnms = (const float*)d_in[19];
    const float* in_proj  = (const float*)d_in[20];
    const float* out_proj = (const float*)d_in[21];
    const float* f_cw = (const float*)d_in[22], *f_cb = (const float*)d_in[23];
    const float* f_xp = (const float*)d_in[24];
    const float* f_dtw = (const float*)d_in[25], *f_dtb = (const float*)d_in[26];
    const float* f_Al = (const float*)d_in[27], *f_D = (const float*)d_in[28];
    const float* b_cw = (const float*)d_in[29], *b_cb = (const float*)d_in[30];
    const float* b_xp = (const float*)d_in[31];
    const float* b_dtw = (const float*)d_in[32], *b_dtb = (const float*)d_in[33];
    const float* b_Al = (const float*)d_in[34], *b_D = (const float*)d_in[35];
    const float* lng = (const float*)d_in[36], *lnb = (const float*)d_in[37];
    const float* w1 = (const float*)d_in[38], *b1 = (const float*)d_in[39];
    const float* w2 = (const float*)d_in[40], *b2 = (const float*)d_in[41];
    const float* w3 = (const float*)d_in[42], *b3 = (const float*)d_in[43];

    float* ws = (float*)d_ws;
    const long NC = (long)NNODE * CDIM;   // 2,097,152

    // Phase A arena (graph attention)
    float* q    = ws + 0 * NC;
    float* kbuf = ws + 1 * NC;
    float* vbuf = ws + 2 * NC;
    float* xr   = ws + 3 * NC;
    float* agg  = ws + 4 * NC;            // becomes hmix in place
    float* tbuf = ws + 5 * NC;
    float* h1   = ws + 6 * NC;            // live until h2 graphnorm
    float* ebuf = ws + 7 * NC;            // E*C = 8*NC, spans [7NC,15NC)
    float* logit = ws + 15 * NC;
    float* eaten = logit + NEDGE;
    float* mbuf  = eaten + NEDGE;
    float* sbuf  = mbuf + NNODE;

    // ---- Graph attention -------------------------------------------------
    launch_gemm(x, wq, bq, q,    NNODE, CDIM, CDIM, stream);
    launch_gemm(x, wk, bk, kbuf, NNODE, CDIM, CDIM, stream);
    launch_gemm(x, wv, bv, vbuf, NNODE, CDIM, CDIM, stream);
    launch_gemm(x, wskip, bskip, xr, NNODE, CDIM, CDIM, stream);
    launch_gemm(eattr, we, be, ebuf, NEDGE, CDIM, EDIMK, stream);   // safe path

    init_attn_kernel<<<(int)(NC / 256), 256, 0, stream>>>(mbuf, sbuf, agg);
    edge_logit_kernel<<<NEDGE * 32 / 256, 256, 0, stream>>>(q, kbuf, ebuf, ei, logit);
    edge_max_kernel<<<NEDGE / 256, 256, 0, stream>>>(logit, ei, mbuf);
    edge_exp_kernel<<<NEDGE / 256, 256, 0, stream>>>(logit, ei, mbuf, eaten, sbuf);
    edge_agg_kernel<<<(int)((long)NEDGE * CDIM / 256), 256, 0, stream>>>(
        vbuf, ebuf, eaten, sbuf, ei, agg);
    beta_mix_kernel<<<NNODE * 32 / 256, 256, 0, stream>>>(agg, xr, wbeta);
    launch_gemm(agg, wlin, blin, tbuf, NNODE, CDIM, CDIM, stream);
    graph_norm_kernel<<<NGRAPH, CDIM, 0, stream>>>(tbuf, x, nullptr, h1, gnw, gnb, gnms);

    // ---- Mamba (both dirs share in_proj/out_proj; flip commutes w/ GEMM) -
    float* xz      = ws + 0 * NC;   // N x 512 (q..xr dead)
    float* xi_f    = ws + 4 * NC;   // N x 256
    float* xi_b    = ws + 7 * NC;   // N x 256 (e dead)
    float* dbc_f   = ws + 9 * NC;   // N x 40
    float* dbc_b   = dbc_f + (long)NNODE * 40;
    float* delta_f = ws + 10 * NC;  // N x 256
    float* delta_b = ws + 12 * NC;
    float* yz_f    = ws + 14 * NC;  // N x 256
    float* yz_b    = ws + 16 * NC;  // high water = 18*NC floats (~151 MB)

    launch_gemm(x, in_proj, nullptr, xz, NNODE, 2 * EDCH, CDIM, stream);
    const int NE = NNODE * EDCH;
    mamba_conv_kernel<<<NE / 256, 256, 0, stream>>>(xz, f_cw, f_cb, xi_f, 0);
    mamba_conv_kernel<<<NE / 256, 256, 0, stream>>>(xz, b_cw, b_cb, xi_b, 1);
    launch_gemm(xi_f, f_xp, nullptr, dbc_f, NNODE, 40, EDCH, stream);   // safe path
    launch_gemm(xi_b, b_xp, nullptr, dbc_b, NNODE, 40, EDCH, stream);   // safe path
    mamba_delta_kernel<<<NE / 256, 256, 0, stream>>>(dbc_f, f_dtw, f_dtb, delta_f);
    mamba_delta_kernel<<<NE / 256, 256, 0, stream>>>(dbc_b, b_dtw, b_dtb, delta_b);
    mamba_scan_kernel<<<NGRAPH * EDCH / 256, 256, 0, stream>>>(
        xi_f, delta_f, dbc_f, xz, f_Al, f_D, yz_f, 0);
    mamba_scan_kernel<<<NGRAPH * EDCH / 256, 256, 0, stream>>>(
        xi_b, delta_b, dbc_b, xz, b_Al, b_D, yz_b, 1);

    float* mix_f  = ws + 0 * NC;    // xz dead after scans
    float* mix_b  = ws + 1 * NC;
    float* hd_ln  = ws + 2 * NC;
    float* outbuf = ws + 3 * NC;
    launch_gemm(yz_f, out_proj, nullptr, mix_f, NNODE, CDIM, EDCH, stream);
    launch_gemm(yz_b, out_proj, nullptr, mix_b, NNODE, CDIM, EDCH, stream);
    combine_ln_kernel<<<NNODE * 32 / 256, 256, 0, stream>>>(x, mix_f, mix_b, lng, lnb, hd_ln);
    // h2 = graphnorm(hd_ln + x); out = h1 + h2 (fused via out_add)
    graph_norm_kernel<<<NGRAPH, CDIM, 0, stream>>>(hd_ln, x, h1, outbuf, gnw, gnb, gnms);

    // ---- MLP + final GraphNorm ------------------------------------------
    float* a1  = ws + 4 * NC;       // N x 512 (h1/xi dead now)
    float* a2  = ws + 8 * NC;       // N x 512
    float* mlp = ws + 12 * NC;      // N x 128
    launch_gemm(outbuf, w1, b1, a1, NNODE, HID, CDIM, stream);
    launch_gemm(outbuf, w2, b2, a2, NNODE, HID, CDIM, stream);
    mlp_gate_kernel<<<(int)((long)NNODE * HID / 256), 256, 0, stream>>>(a1, a2);
    launch_gemm(a1, w3, b3, mlp, NNODE, CDIM, HID, stream);
    // final = graphnorm(outbuf + mlp)
    graph_norm_kernel<<<NGRAPH, CDIM, 0, stream>>>(mlp, outbuf, nullptr, (float*)d_out,
                                                   gnw, gnb, gnms);
}
